// CUTSEncoder_16226386444786
// MI455X (gfx1250) — compile-verified, matrix operand order checked by reading, not run on hardware
//
#include <hip/hip_runtime.h>
#include <string.h>

// ---------- types ----------
typedef __attribute__((ext_vector_type(16))) __bf16 v16bf;
typedef __attribute__((ext_vector_type(8)))  float  v8f;

struct __attribute__((aligned(16))) U4 { unsigned int x, y, z, w; };
union BFrag { v16bf v; U4 q[2]; };

__device__ __forceinline__ unsigned short f32_to_bf16(float f) {
    unsigned int u = __float_as_uint(f);
    u = u + 0x7fffu + ((u >> 16) & 1u);   // round-to-nearest-even
    return (unsigned short)(u >> 16);
}
__device__ __forceinline__ float bf16_to_f32(unsigned short h) {
    return __uint_as_float(((unsigned int)h) << 16);
}

#define BB 16
#define HH 256
#define WW 256
#define NPIX (BB * HH * WW)          // 1048576
#define INVN (1.0f / 1048576.0f)

// ---------- 0: zero stats ----------
__global__ void cuts_zero_stats(float* s) {
    for (int i = threadIdx.x; i < 896; i += 256) s[i] = 0.0f;
}

// ---------- 1: pack conv2/conv3 weights into per-lane WMMA B fragments ----------
// frag index f = (tap*KS + ks)*NT + nt ; element (lane,e):
//   n = nt*16 + (lane&15);  K = e + (e>=8 ? 8:0) + (lane>=16 ? 8:0);  c_in = ks*32 + K
__global__ void cuts_prep_weights(const float* __restrict__ w2, const float* __restrict__ w3,
                                  unsigned short* __restrict__ f2, unsigned short* __restrict__ f3) {
    int idx = blockIdx.x * 256 + threadIdx.x;
    const int N2 = 9 * 4 * 512;         // conv2: NT=4, KS=1
    const int N3 = 9 * 2 * 8 * 512;     // conv3: NT=8, KS=2
    if (idx < N2) {
        int e = idx & 15, lane = (idx >> 4) & 31, f = idx >> 9;
        int nt = f & 3, tap = f >> 2;
        int K = e + (e >= 8 ? 8 : 0) + (lane >= 16 ? 8 : 0);
        int o = nt * 16 + (lane & 15);
        int ky = tap / 3, kx = tap % 3;
        f2[idx] = f32_to_bf16(w2[((o * 32 + K) * 3 + ky) * 3 + kx]);
    } else if (idx < N2 + N3) {
        int j = idx - N2;
        int e = j & 15, lane = (j >> 4) & 31, f = j >> 9;
        int nt = f & 7, ks = (f >> 3) & 1, tap = f >> 4;
        int K = e + (e >= 8 ? 8 : 0) + (lane >= 16 ? 8 : 0);
        int cin = ks * 32 + K;
        int o = nt * 16 + (lane & 15);
        int ky = tap / 3, kx = tap % 3;
        f3[j] = f32_to_bf16(w3[((o * 64 + cin) * 3 + ky) * 3 + kx]);
    }
}

// ---------- 2: conv1 direct (C=3, K=27), NCHW f32 -> NHWC bf16 pre-BN, + stats ----------
__global__ __launch_bounds__(256) void cuts_conv1(const float* __restrict__ x,
                                                  const float* __restrict__ w,
                                                  const float* __restrict__ bias,
                                                  unsigned short* __restrict__ y1,
                                                  float* __restrict__ ssum, float* __restrict__ ssq) {
    __shared__ float ls[64];
    int tid = threadIdx.x;
    if (tid < 64) ls[tid] = 0.0f;
    __syncthreads();
    int c = tid & 31;
    long pos = (long)blockIdx.x * 8 + (tid >> 5);
    int wv = (int)(pos & 255), hv = (int)((pos >> 8) & 255), b = (int)(pos >> 16);
    float acc = bias[c];
    #pragma unroll
    for (int ci = 0; ci < 3; ++ci)
        #pragma unroll
        for (int dy = 0; dy < 3; ++dy) {
            int ih = hv + dy - 1;
            if ((unsigned)ih >= 256u) continue;
            #pragma unroll
            for (int dx = 0; dx < 3; ++dx) {
                int iw = wv + dx - 1;
                if ((unsigned)iw >= 256u) continue;
                acc += x[(((size_t)b * 3 + ci) * 256 + ih) * 256 + iw] *
                       w[((c * 3 + ci) * 3 + dy) * 3 + dx];
            }
        }
    y1[(size_t)pos * 32 + c] = f32_to_bf16(acc);
    atomicAdd(&ls[c], acc);
    atomicAdd(&ls[32 + c], acc * acc);
    __syncthreads();
    if (tid < 32) { atomicAdd(&ssum[tid], ls[tid]); atomicAdd(&ssq[tid], ls[32 + tid]); }
}

// ---------- 3: BN finalize: per-channel scale a, shift b ----------
__global__ void cuts_bn_finalize(const float* __restrict__ ssum, const float* __restrict__ ssq,
                                 const float* __restrict__ gamma, const float* __restrict__ beta,
                                 float* __restrict__ a, float* __restrict__ bc, int C) {
    int c = blockIdx.x * blockDim.x + threadIdx.x;
    if (c >= C) return;
    float mean = ssum[c] * INVN;
    float var  = ssq[c] * INVN - mean * mean;
    float inv  = rsqrtf(var + 1e-5f);
    float av = gamma[c] * inv;
    a[c]  = av;
    bc[c] = beta[c] - mean * av;
}

// ---------- 4: BN + LeakyReLU apply in place on bf16 NHWC ----------
__global__ __launch_bounds__(256) void cuts_bn_apply(unsigned short* __restrict__ y,
                                                     const float* __restrict__ a,
                                                     const float* __restrict__ bc,
                                                     int cmask, size_t total8) {
    size_t i = (size_t)blockIdx.x * 256 + threadIdx.x;
    if (i >= total8) return;
    unsigned short* p = y + i * 8;
    U4 q = *(const U4*)p;
    unsigned int wrd[4] = {q.x, q.y, q.z, q.w};
    int c0 = (int)((i * 8) & (size_t)cmask);
    unsigned int out[4];
    #pragma unroll
    for (int k = 0; k < 4; ++k) {
        float v0 = bf16_to_f32((unsigned short)(wrd[k] & 0xffffu));
        float v1 = bf16_to_f32((unsigned short)(wrd[k] >> 16));
        int cc = c0 + 2 * k;
        float y0 = a[cc] * v0 + bc[cc];
        float y1v = a[cc + 1] * v1 + bc[cc + 1];
        y0  = y0  >= 0.0f ? y0  : 0.01f * y0;
        y1v = y1v >= 0.0f ? y1v : 0.01f * y1v;
        out[k] = (unsigned int)f32_to_bf16(y0) | ((unsigned int)f32_to_bf16(y1v) << 16);
    }
    U4 r = {out[0], out[1], out[2], out[3]};
    *(U4*)p = r;
}

// ---------- 5: conv2 implicit GEMM via WMMA bf16 (32->64 ch) ----------
__global__ __launch_bounds__(128) void cuts_conv2_wmma(const unsigned short* __restrict__ z1,
                                                       const unsigned short* __restrict__ wfrag,
                                                       const float* __restrict__ bias,
                                                       unsigned short* __restrict__ y2,
                                                       float* __restrict__ ssum, float* __restrict__ ssq) {
    __shared__ unsigned short lds[3 * 66 * 32];
    int wt = blockIdx.x, h = blockIdx.y, b = blockIdx.z;
    int w0 = wt * 64;
    int tid = threadIdx.x;
    for (int p = tid; p < 3 * 66; p += 128) {
        int r = p / 66, col = p % 66;
        int gh = h + r - 1, gw = w0 + col - 1;
        U4* dst = (U4*)&lds[p * 32];
        if ((unsigned)gh < 256u && (unsigned)gw < 256u) {
            const U4* src = (const U4*)&z1[(((size_t)b * 256 + gh) * 256 + gw) * 32];
            dst[0] = src[0]; dst[1] = src[1]; dst[2] = src[2]; dst[3] = src[3];
        } else {
            U4 zz = {0, 0, 0, 0};
            dst[0] = zz; dst[1] = zz; dst[2] = zz; dst[3] = zz;
        }
    }
    __syncthreads();

    int wave = tid >> 5;          // n-tile 0..3
    int lane = tid & 31;
    int m = lane & 15;
    int hi = lane >> 4;
    int koff = hi * 8;            // channel-chunk offset per A/B lane layout

    v8f acc[4];
    #pragma unroll
    for (int mt = 0; mt < 4; ++mt) { v8f z = {}; acc[mt] = z; }

    for (int tap = 0; tap < 9; ++tap) {
        int dy = tap / 3, dx = tap % 3;
        BFrag bf;
        const U4* bp = (const U4*)&wfrag[(size_t)((tap * 4 + wave) * 32 + lane) * 16];
        bf.q[0] = bp[0]; bf.q[1] = bp[1];
        #pragma unroll
        for (int mt = 0; mt < 4; ++mt) {
            int col = mt * 16 + m + dx;                 // -1 pad already folded into LDS slab
            const unsigned short* ap = &lds[(dy * 66 + col) * 32 + koff];
            BFrag af;
            af.q[0] = *(const U4*)(ap);
            af.q[1] = *(const U4*)(ap + 16);
            acc[mt] = __builtin_amdgcn_wmma_f32_16x16x32_bf16(
                false, af.v, false, bf.v, (short)0, acc[mt], false, false);
        }
    }

    int n = wave * 16 + m;
    float bv = bias[n];
    float psum = 0.0f, psq = 0.0f;
    #pragma unroll
    for (int mt = 0; mt < 4; ++mt) {
        #pragma unroll
        for (int v = 0; v < 8; ++v) {
            float val = acc[mt][v] + bv;
            psum += val; psq += val * val;
            int wpos = w0 + mt * 16 + v + hi * 8;       // D: M = v + 8*hi
            y2[(((size_t)b * 256 + h) * 256 + wpos) * 64 + n] = f32_to_bf16(val);
        }
    }
    psum += __shfl_xor(psum, 16, 32);
    psq  += __shfl_xor(psq, 16, 32);
    if (hi == 0) { atomicAdd(&ssum[n], psum); atomicAdd(&ssq[n], psq); }
}

// ---------- 6: conv3 implicit GEMM via WMMA bf16 (64->128 ch), stats only ----------
__global__ __launch_bounds__(256) void cuts_conv3_wmma(const unsigned short* __restrict__ z2,
                                                       const unsigned short* __restrict__ wfrag,
                                                       const float* __restrict__ bias,
                                                       float* __restrict__ ssum, float* __restrict__ ssq) {
    __shared__ unsigned short lds[3 * 66 * 64];
    int wt = blockIdx.x, h = blockIdx.y, b = blockIdx.z;
    int w0 = wt * 64;
    int tid = threadIdx.x;
    for (int p = tid; p < 3 * 66; p += 256) {
        int r = p / 66, col = p % 66;
        int gh = h + r - 1, gw = w0 + col - 1;
        U4* dst = (U4*)&lds[p * 64];
        if ((unsigned)gh < 256u && (unsigned)gw < 256u) {
            const U4* src = (const U4*)&z2[(((size_t)b * 256 + gh) * 256 + gw) * 64];
            #pragma unroll
            for (int k = 0; k < 8; ++k) dst[k] = src[k];
        } else {
            U4 zz = {0, 0, 0, 0};
            #pragma unroll
            for (int k = 0; k < 8; ++k) dst[k] = zz;
        }
    }
    __syncthreads();

    int wave = tid >> 5;          // n-tile 0..7
    int lane = tid & 31;
    int m = lane & 15;
    int hi = lane >> 4;
    int koff = hi * 8;

    v8f acc[4];
    #pragma unroll
    for (int mt = 0; mt < 4; ++mt) { v8f z = {}; acc[mt] = z; }

    for (int tap = 0; tap < 9; ++tap) {
        int dy = tap / 3, dx = tap % 3;
        for (int ks = 0; ks < 2; ++ks) {
            BFrag bf;
            const U4* bp = (const U4*)&wfrag[(size_t)(((tap * 2 + ks) * 8 + wave) * 32 + lane) * 16];
            bf.q[0] = bp[0]; bf.q[1] = bp[1];
            #pragma unroll
            for (int mt = 0; mt < 4; ++mt) {
                int col = mt * 16 + m + dx;
                const unsigned short* ap = &lds[(dy * 66 + col) * 64 + ks * 32 + koff];
                BFrag af;
                af.q[0] = *(const U4*)(ap);
                af.q[1] = *(const U4*)(ap + 16);
                acc[mt] = __builtin_amdgcn_wmma_f32_16x16x32_bf16(
                    false, af.v, false, bf.v, (short)0, acc[mt], false, false);
            }
        }
    }

    int n = wave * 16 + m;
    float bv = bias[n];
    float psum = 0.0f, psq = 0.0f;
    #pragma unroll
    for (int mt = 0; mt < 4; ++mt)
        #pragma unroll
        for (int v = 0; v < 8; ++v) {
            float val = acc[mt][v] + bv;
            psum += val; psq += val * val;
        }
    psum += __shfl_xor(psum, 16, 32);
    psq  += __shfl_xor(psq, 16, 32);
    if (hi == 0) { atomicAdd(&ssum[n], psum); atomicAdd(&ssq[n], psq); }
}

// ---------- 7: gathered outputs: z_anchors / z_positives (conv3@points + BN3 + lrelu), x_anchors ----------
__global__ __launch_bounds__(160) void cuts_gather(const float* __restrict__ x,
                                                   const int* __restrict__ anchors,
                                                   const int* __restrict__ positives,
                                                   const unsigned short* __restrict__ z2,
                                                   const float* __restrict__ w3,
                                                   const float* __restrict__ b3,
                                                   const float* __restrict__ a3,
                                                   const float* __restrict__ bc3,
                                                   float* __restrict__ out) {
    const int XR = 2352, ZA = 4704, ZP = 6752;
    int blk = blockIdx.x, tid = threadIdx.x;
    (void)XR;
    if (blk < 32) {
        int b = blk & 15;
        const int* src = (blk < 16) ? anchors : positives;
        int h = src[b * 8 + 6], w = src[b * 8 + 7];
        if (tid >= 128) return;
        int c = tid;
        float acc = b3[c];
        for (int ci = 0; ci < 64; ++ci)
            #pragma unroll
            for (int ky = 0; ky < 3; ++ky) {
                int ih = h + ky - 1;
                #pragma unroll
                for (int kx = 0; kx < 3; ++kx) {
                    int iw = w + kx - 1;
                    float zv = ((unsigned)ih < 256u && (unsigned)iw < 256u)
                        ? bf16_to_f32(z2[(((size_t)b * 256 + ih) * 256 + iw) * 64 + ci]) : 0.0f;
                    acc += zv * w3[((c * 64 + ci) * 3 + ky) * 3 + kx];
                }
            }
        float yv = a3[c] * acc + bc3[c];
        yv = yv >= 0.0f ? yv : 0.01f * yv;
        out[(blk < 16 ? ZA : ZP) + b * 128 + c] = yv;
    } else {
        int b = blk - 32;
        if (tid >= 147) return;
        int h = anchors[b * 8 + 6], w = anchors[b * 8 + 7];
        int ci = tid / 49, rem = tid % 49, py = rem / 7, px = rem % 7;
        out[b * 147 + tid] = x[(((size_t)b * 3 + ci) * 256 + (h - 3 + py)) * 256 + (w - 3 + px)];
    }
}

// ---------- 8: x_recon = z_anchors @ recon_w^T + recon_b ----------
__global__ __launch_bounds__(160) void cuts_recon(const float* __restrict__ out_in,
                                                  const float* __restrict__ rw,
                                                  const float* __restrict__ rb,
                                                  float* __restrict__ out) {
    __shared__ float zs[128];
    int b = blockIdx.x, tid = threadIdx.x;
    if (tid < 128) zs[tid] = out_in[4704 + b * 128 + tid];
    __syncthreads();
    if (tid >= 147) return;
    float acc = rb[tid];
    const float* wrow = &rw[tid * 128];
    for (int k = 0; k < 128; ++k) acc += zs[k] * wrow[k];
    out[2352 + b * 147 + tid] = acc;
}

// ---------- launch ----------
extern "C" void kernel_launch(void* const* d_in, const int* in_sizes, int n_in,
                              void* d_out, int out_size, void* d_ws, size_t ws_size,
                              hipStream_t stream) {
    (void)in_sizes; (void)n_in; (void)out_size; (void)ws_size;
    const float* x      = (const float*)d_in[0];
    const int*   anc    = (const int*)d_in[1];
    const int*   pos    = (const int*)d_in[2];
    const float* w1     = (const float*)d_in[3];
    const float* b1c    = (const float*)d_in[4];
    const float* g1     = (const float*)d_in[5];
    const float* be1    = (const float*)d_in[6];
    const float* w2     = (const float*)d_in[7];
    const float* b2c    = (const float*)d_in[8];
    const float* g2     = (const float*)d_in[9];
    const float* be2    = (const float*)d_in[10];
    const float* w3     = (const float*)d_in[11];
    const float* b3c    = (const float*)d_in[12];
    const float* g3     = (const float*)d_in[13];
    const float* be3    = (const float*)d_in[14];
    const float* rw     = (const float*)d_in[15];
    const float* rb     = (const float*)d_in[16];
    float* out = (float*)d_out;

    // workspace layout
    unsigned short* y1 = (unsigned short*)d_ws;                                   // 33554432 elems
    unsigned short* y2 = (unsigned short*)((char*)d_ws + 67108864);               // 67108864 elems
    float* st = (float*)((char*)d_ws + 67108864 + 134217728);                     // 896 floats
    float *s1 = st,       *q1 = st + 32;
    float *s2 = st + 64,  *q2 = st + 128;
    float *s3 = st + 192, *q3 = st + 320;
    float *a1 = st + 448, *bc1 = st + 480;
    float *a2 = st + 512, *bc2 = st + 576;
    float *a3 = st + 640, *bc3 = st + 768;
    unsigned short* f2 = (unsigned short*)((char*)st + 4096);                     // 18432 elems
    unsigned short* f3 = f2 + 18432;                                             // 73728 elems

    cuts_zero_stats<<<1, 256, 0, stream>>>(st);
    cuts_prep_weights<<<(92160 + 255) / 256, 256, 0, stream>>>(w2, w3, f2, f3);

    cuts_conv1<<<NPIX / 8, 256, 0, stream>>>(x, w1, b1c, y1, s1, q1);
    cuts_bn_finalize<<<1, 128, 0, stream>>>(s1, q1, g1, be1, a1, bc1, 32);
    cuts_bn_apply<<<(NPIX * 32 / 8) / 256, 256, 0, stream>>>(y1, a1, bc1, 31, (size_t)NPIX * 32 / 8);

    cuts_conv2_wmma<<<dim3(4, 256, 16), 128, 0, stream>>>(y1, f2, b2c, y2, s2, q2);
    cuts_bn_finalize<<<1, 128, 0, stream>>>(s2, q2, g2, be2, a2, bc2, 64);
    cuts_bn_apply<<<(NPIX * 64 / 8) / 256, 256, 0, stream>>>(y2, a2, bc2, 63, (size_t)NPIX * 64 / 8);

    cuts_conv3_wmma<<<dim3(4, 256, 16), 256, 0, stream>>>(y2, f3, b3c, s3, q3);
    cuts_bn_finalize<<<1, 128, 0, stream>>>(s3, q3, g3, be3, a3, bc3, 128);

    cuts_gather<<<48, 160, 0, stream>>>(x, anc, pos, y2, w3, b3c, a3, bc3, out);
    cuts_recon<<<16, 160, 0, stream>>>(out, rw, rb, out);
}
